// GAT_7730941133127
// MI455X (gfx1250) — compile-verified, matrix operand order checked by reading, not run on hardware
//
#include <hip/hip_runtime.h>

// ---------------------------------------------------------------------------
// GAT forward for MI455X (gfx1250, wave32, WMMA bf16 w/ f32 accumulate)
// B=4, N=2048, NFEAT=128, NHID=64, NCLASS=32, NHEADS=8
// ---------------------------------------------------------------------------

#define ALPHA_ 0.2f
constexpr int Bc = 4, Nn = 2048, NF = 128, NHID = 64, NCLS = 32, NH = 8;

typedef __attribute__((ext_vector_type(16))) __bf16 v16bf;
typedef __attribute__((ext_vector_type(8)))  float  v8f;

union Frag16 {              // one wave32 16-bit A/B WMMA operand (16 bf16 / lane)
    v16bf  v;
    uint4  q[2];
    __bf16 h[16];
};
union Pack8 {               // 8 bf16 packed for one 16B store
    uint4  q;
    __bf16 h[8];
};

__device__ __forceinline__ v8f wmma_bf16(v16bf a, v16bf b, v8f c) {
    // D = A(16x32 bf16) x B(32x16 bf16) + C(16x16 f32)
    return __builtin_amdgcn_wmma_f32_16x16x32_bf16(false, a, false, b,
                                                   (short)0, c, false, false);
}

__device__ __forceinline__ float eluf(float x) {
    return x > 0.f ? x : (__expf(x) - 1.f);
}

// ---------------------------------------------------------------------------
// Prep kernels
// ---------------------------------------------------------------------------

// adj (int32, B*N*N) -> bitmask words (B*N*(N/32)); bit j%32 of word j/32
__global__ void k_pack_adj(const int* __restrict__ adj,
                           unsigned* __restrict__ bits, int nwords) {
    int t = blockIdx.x * blockDim.x + threadIdx.x;
    if (t >= nwords) return;
    const int* p = adj + (size_t)t * 32;
    unsigned m = 0;
#pragma unroll
    for (int j = 0; j < 32; ++j) m |= (p[j] > 0 ? 1u : 0u) << j;
    bits[t] = m;
}

__global__ void k_cvt_bf16(const float* __restrict__ src,
                           __bf16* __restrict__ dst, int n) {
    int t = blockIdx.x * blockDim.x + threadIdx.x;
    if (t < n) dst[t] = (__bf16)src[t];
}

// W [H,128,64] f32 -> WT [H,64,128] bf16
__global__ void k_wt(const float* __restrict__ W, __bf16* __restrict__ WT) {
    int t = blockIdx.x * blockDim.x + threadIdx.x;   // H*64*128 = 65536
    int hh = t >> 13;
    int r  = t & 8191;
    int d  = r >> 7;
    int f  = r & 127;
    WT[t] = (__bf16)W[((size_t)hh * NF + f) * NHID + d];
}

// Wo [512,32] f32 -> WoT [32,512] bf16
__global__ void k_wot(const float* __restrict__ Wo, __bf16* __restrict__ WoT) {
    int t = blockIdx.x * blockDim.x + threadIdx.x;   // 32*512 = 16384
    int c = t >> 9;
    int k = t & 511;
    WoT[t] = (__bf16)Wo[(size_t)k * NCLS + c];
}

// ---------------------------------------------------------------------------
// Phase 1: WhT[b*8+h][d][n] = (x[b] @ W[h])^T   via WMMA
// grid = B*H*(N/16) = 4096 blocks x 32 threads (1 wave)
// ---------------------------------------------------------------------------
__global__ void k_wh(const __bf16* __restrict__ xb,   // [B,N,128]
                     const __bf16* __restrict__ WT,   // [H,64,128]
                     __bf16* __restrict__ WhT) {      // [B*H,64,N]
    const int bh   = blockIdx.x >> 7;
    const int i0   = (blockIdx.x & 127) << 4;
    const int hh   = bh & 7;
    const int b    = bh >> 3;
    const int lane = threadIdx.x;
    const int row  = lane & 15;
    const int koff = (lane >> 4) << 3;

    const __bf16* ar = xb + (size_t)(b * Nn + i0 + row) * NF + koff;
    const __bf16* wr = WT + ((size_t)hh * NHID + row) * NF + koff;

    v8f acc[4] = {};
#pragma unroll
    for (int k0 = 0; k0 < NF; k0 += 32) {
        Frag16 a;
        a.q[0] = *(const uint4*)(ar + k0);
        a.q[1] = *(const uint4*)(ar + k0 + 16);
#pragma unroll
        for (int c = 0; c < 4; ++c) {
            Frag16 bf;
            const __bf16* bp = wr + (size_t)c * 16 * NF + k0;
            bf.q[0] = *(const uint4*)(bp);
            bf.q[1] = *(const uint4*)(bp + 16);
            acc[c] = wmma_bf16(a.v, bf.v, acc[c]);
        }
    }
    const int mb = (lane >> 4) << 3;
#pragma unroll
    for (int c = 0; c < 4; ++c) {
        Pack8 s;
#pragma unroll
        for (int k = 0; k < 8; ++k) s.h[k] = (__bf16)acc[c][k];
        *(uint4*)(WhT + ((size_t)bh * NHID + c * 16 + row) * Nn + i0 + mb) = s.q;
    }
}

// f1/f2 [B*H, N]: per-node dot of WhT column with a1/a2
__global__ void k_f1f2(const __bf16* __restrict__ WhT,
                       const float* __restrict__ a1, const float* __restrict__ a2,
                       float* __restrict__ f1, float* __restrict__ f2) {
    int t  = blockIdx.x * blockDim.x + threadIdx.x;   // B*H*N = 65536
    int bh = t >> 11;
    int n  = t & 2047;
    int hh = bh & 7;
    const __bf16* w  = WhT + (size_t)bh * NHID * Nn + n;
    const float*  A1 = a1 + hh * NHID;
    const float*  A2 = a2 + hh * NHID;
    float s1 = 0.f, s2 = 0.f;
#pragma unroll 8
    for (int d = 0; d < NHID; ++d) {
        float wv = (float)w[(size_t)d * Nn];
        s1 += wv * A1[d];
        s2 += wv * A2[d];
    }
    f1[t] = s1;
    f2[t] = s2;
}

// ---------------------------------------------------------------------------
// Phase 2: fused masked-softmax attention, all 8 heads per workgroup.
// grid = B*(N/16) = 512 blocks x 256 threads; wave w handles head w.
// h[b][n][head*64+d] = elu( softmax_j(mask? lrelu(f1_i+f2_j):-inf) @ Wh )
// ---------------------------------------------------------------------------
__global__ void k_attn1(const float* __restrict__ f1, const float* __restrict__ f2,
                        const unsigned* __restrict__ adjbits,   // [B,N,N/32]
                        const __bf16* __restrict__ WhT,         // [B*H,64,N]
                        __bf16* __restrict__ hout) {            // [B,N,H*64]
    const int b    = blockIdx.x >> 7;
    const int i0   = (blockIdx.x & 127) << 4;
    const int head = threadIdx.x >> 5;
    const int lane = threadIdx.x & 31;
    const int row  = lane & 15;
    const int koff = (lane >> 4) << 3;
    const int bh   = b * NH + head;

    const float      f1v  = f1[bh * Nn + i0 + row];
    const float*     f2r  = f2 + (size_t)bh * Nn;
    const unsigned*  mrow = adjbits + ((size_t)(b * Nn + i0 + row) << 6);
    const __bf16*    whb  = WhT + (size_t)bh * NHID * Nn;

    v8f acc0 = {}, acc1 = {}, acc2 = {}, acc3 = {};
    float psum = 0.f;

    for (int j0 = 0; j0 < Nn; j0 += 32) {
        const unsigned m  = mrow[j0 >> 5];
        const float*   fp = f2r + j0 + koff;
        float f2v[16];
        *(float4*)(&f2v[0])  = *(const float4*)(fp);
        *(float4*)(&f2v[4])  = *(const float4*)(fp + 4);
        *(float4*)(&f2v[8])  = *(const float4*)(fp + 16);
        *(float4*)(&f2v[12]) = *(const float4*)(fp + 20);

        Frag16 a;
#pragma unroll
        for (int e = 0; e < 16; ++e) {
            const int k = koff + (e & 7) + ((e >> 3) << 4);  // col within j-block
            float ev = f1v + f2v[e];
            ev = ev > 0.f ? ev : ALPHA_ * ev;
            float p = ((m >> k) & 1u) ? __expf(ev) : 0.f;
            psum += p;
            a.h[e] = (__bf16)p;
        }

        const __bf16* bp0 = whb + (size_t)row * Nn + j0 + koff;
        Frag16 bf;
        bf.q[0] = *(const uint4*)(bp0);
        bf.q[1] = *(const uint4*)(bp0 + 16);
        acc0 = wmma_bf16(a.v, bf.v, acc0);
        bf.q[0] = *(const uint4*)(bp0 + 16 * Nn);
        bf.q[1] = *(const uint4*)(bp0 + 16 * Nn + 16);
        acc1 = wmma_bf16(a.v, bf.v, acc1);
        bf.q[0] = *(const uint4*)(bp0 + 32 * Nn);
        bf.q[1] = *(const uint4*)(bp0 + 32 * Nn + 16);
        acc2 = wmma_bf16(a.v, bf.v, acc2);
        bf.q[0] = *(const uint4*)(bp0 + 48 * Nn);
        bf.q[1] = *(const uint4*)(bp0 + 48 * Nn + 16);
        acc3 = wmma_bf16(a.v, bf.v, acc3);
    }

    const float den = psum + __shfl_xor(psum, 16);   // full row sum, row = lane&15
    const int   mb  = (lane >> 4) << 3;
    float dinv[8];
#pragma unroll
    for (int k = 0; k < 8; ++k)
        dinv[k] = 1.f / fmaxf(__shfl(den, k + mb), 1e-30f);

    __bf16* ho = hout + (size_t)(b * Nn + i0 + mb) * (NH * NHID) + head * NHID + row;
#pragma unroll
    for (int k = 0; k < 8; ++k) {
        __bf16* p = ho + (size_t)k * (NH * NHID);
        p[0]  = (__bf16)eluf(acc0[k] * dinv[k]);
        p[16] = (__bf16)eluf(acc1[k] * dinv[k]);
        p[32] = (__bf16)eluf(acc2[k] * dinv[k]);
        p[48] = (__bf16)eluf(acc3[k] * dinv[k]);
    }
}

// ---------------------------------------------------------------------------
// Phase 3a: Wh2 = h @ Wo  (K = 512) via WMMA.  grid = B*(N/16) = 512 x 32
// ---------------------------------------------------------------------------
__global__ void k_wh2(const __bf16* __restrict__ hb,    // [B,N,512]
                      const __bf16* __restrict__ WoT,   // [32,512]
                      float* __restrict__ Wh2,          // [B,N,32]
                      __bf16* __restrict__ Wh2T) {      // [B,32,N]
    const int b    = blockIdx.x >> 7;
    const int i0   = (blockIdx.x & 127) << 4;
    const int lane = threadIdx.x;
    const int row  = lane & 15;
    const int koff = (lane >> 4) << 3;

    const __bf16* ar = hb + (size_t)(b * Nn + i0 + row) * (NH * NHID) + koff;
    const __bf16* wr = WoT + (size_t)row * (NH * NHID) + koff;

    v8f acc[2] = {};
    for (int k0 = 0; k0 < NH * NHID; k0 += 32) {
        Frag16 a;
        a.q[0] = *(const uint4*)(ar + k0);
        a.q[1] = *(const uint4*)(ar + k0 + 16);
#pragma unroll
        for (int c = 0; c < 2; ++c) {
            Frag16 bf;
            const __bf16* bp = wr + (size_t)c * 16 * (NH * NHID) + k0;
            bf.q[0] = *(const uint4*)(bp);
            bf.q[1] = *(const uint4*)(bp + 16);
            acc[c] = wmma_bf16(a.v, bf.v, acc[c]);
        }
    }
    const int mb = (lane >> 4) << 3;
#pragma unroll
    for (int c = 0; c < 2; ++c) {
        Pack8 s;
#pragma unroll
        for (int k = 0; k < 8; ++k) {
            s.h[k] = (__bf16)acc[c][k];
            Wh2[(size_t)(b * Nn + i0 + mb + k) * NCLS + c * 16 + row] = acc[c][k];
        }
        *(uint4*)(Wh2T + ((size_t)b * NCLS + c * 16 + row) * Nn + i0 + mb) = s.q;
    }
}

// g1/g2 [B,N]: dot of Wh2 row with ao1/ao2
__global__ void k_g(const float* __restrict__ Wh2,
                    const float* __restrict__ ao1, const float* __restrict__ ao2,
                    float* __restrict__ g1, float* __restrict__ g2) {
    int t = blockIdx.x * blockDim.x + threadIdx.x;   // B*N = 8192
    const float* r = Wh2 + (size_t)t * NCLS;
    float s1 = 0.f, s2 = 0.f;
#pragma unroll
    for (int c = 0; c < NCLS; ++c) {
        s1 += r[c] * ao1[c];
        s2 += r[c] * ao2[c];
    }
    g1[t] = s1;
    g2[t] = s2;
}

// ---------------------------------------------------------------------------
// Phase 3b: output attention (no ELU).  grid = 64 blocks x 256 (8 waves),
// wave w handles row-block blockIdx*8+w of B*(N/16)=512 row-blocks.
// ---------------------------------------------------------------------------
__global__ void k_attn2(const float* __restrict__ g1, const float* __restrict__ g2,
                        const unsigned* __restrict__ adjbits,
                        const __bf16* __restrict__ Wh2T,   // [B,32,N]
                        float* __restrict__ out) {         // [B,N,32]
    const int ib   = blockIdx.x * 8 + (threadIdx.x >> 5);
    const int b    = ib >> 7;
    const int i0   = (ib & 127) << 4;
    const int lane = threadIdx.x & 31;
    const int row  = lane & 15;
    const int koff = (lane >> 4) << 3;

    const float     g1v  = g1[b * Nn + i0 + row];
    const float*    g2r  = g2 + (size_t)b * Nn;
    const unsigned* mrow = adjbits + ((size_t)(b * Nn + i0 + row) << 6);
    const __bf16*   wb   = Wh2T + (size_t)b * NCLS * Nn;

    v8f acc0 = {}, acc1 = {};
    float psum = 0.f;

    for (int j0 = 0; j0 < Nn; j0 += 32) {
        const unsigned m  = mrow[j0 >> 5];
        const float*   fp = g2r + j0 + koff;
        float f2v[16];
        *(float4*)(&f2v[0])  = *(const float4*)(fp);
        *(float4*)(&f2v[4])  = *(const float4*)(fp + 4);
        *(float4*)(&f2v[8])  = *(const float4*)(fp + 16);
        *(float4*)(&f2v[12]) = *(const float4*)(fp + 20);

        Frag16 a;
#pragma unroll
        for (int e = 0; e < 16; ++e) {
            const int k = koff + (e & 7) + ((e >> 3) << 4);
            float ev = g1v + f2v[e];
            ev = ev > 0.f ? ev : ALPHA_ * ev;
            float p = ((m >> k) & 1u) ? __expf(ev) : 0.f;
            psum += p;
            a.h[e] = (__bf16)p;
        }

        const __bf16* bp0 = wb + (size_t)row * Nn + j0 + koff;
        Frag16 bf;
        bf.q[0] = *(const uint4*)(bp0);
        bf.q[1] = *(const uint4*)(bp0 + 16);
        acc0 = wmma_bf16(a.v, bf.v, acc0);
        bf.q[0] = *(const uint4*)(bp0 + 16 * Nn);
        bf.q[1] = *(const uint4*)(bp0 + 16 * Nn + 16);
        acc1 = wmma_bf16(a.v, bf.v, acc1);
    }

    const float den = psum + __shfl_xor(psum, 16);
    const int   mb  = (lane >> 4) << 3;
    float* orow = out + (size_t)(b * Nn + i0 + mb) * NCLS + row;
#pragma unroll
    for (int k = 0; k < 8; ++k) {
        const float dinv = 1.f / fmaxf(__shfl(den, k + mb), 1e-30f);
        orow[(size_t)k * NCLS + 0]  = acc0[k] * dinv;
        orow[(size_t)k * NCLS + 16] = acc1[k] * dinv;
    }
}

// ---------------------------------------------------------------------------
// Launch
// ---------------------------------------------------------------------------
extern "C" void kernel_launch(void* const* d_in, const int* in_sizes, int n_in,
                              void* d_out, int out_size, void* d_ws, size_t ws_size,
                              hipStream_t stream) {
    (void)in_sizes; (void)n_in; (void)out_size; (void)ws_size;

    const float* x   = (const float*)d_in[0];   // [B,N,128]
    const int*   adj = (const int*)d_in[1];     // [B,N,N]
    const float* W   = (const float*)d_in[2];   // [H,128,64]
    const float* a1  = (const float*)d_in[3];   // [H,64]
    const float* a2  = (const float*)d_in[4];   // [H,64]
    const float* Wo  = (const float*)d_in[5];   // [512,32]
    const float* ao1 = (const float*)d_in[6];   // [32]
    const float* ao2 = (const float*)d_in[7];   // [32]
    float* out = (float*)d_out;                 // [B,N,32]

    char* wsp = (char*)d_ws;
    auto alloc = [&](size_t bytes) {
        char* p = wsp;
        wsp += (bytes + 255) & ~(size_t)255;
        return p;
    };
    unsigned* adjbits = (unsigned*)alloc((size_t)Bc * Nn * (Nn / 32) * 4); // 2 MB
    __bf16*   xb      = (__bf16*)  alloc((size_t)Bc * Nn * NF * 2);        // 2 MB
    __bf16*   WT      = (__bf16*)  alloc((size_t)NH * NHID * NF * 2);      // 128 KB
    __bf16*   WoT     = (__bf16*)  alloc((size_t)NCLS * NH * NHID * 2);    // 32 KB
    __bf16*   WhT     = (__bf16*)  alloc((size_t)Bc * NH * NHID * Nn * 2); // 8 MB
    float*    f1      = (float*)   alloc((size_t)Bc * NH * Nn * 4);
    float*    f2      = (float*)   alloc((size_t)Bc * NH * Nn * 4);
    __bf16*   hb      = (__bf16*)  alloc((size_t)Bc * Nn * NH * NHID * 2); // 8 MB
    float*    Wh2     = (float*)   alloc((size_t)Bc * Nn * NCLS * 4);      // 1 MB
    __bf16*   Wh2T    = (__bf16*)  alloc((size_t)Bc * NCLS * Nn * 2);
    float*    g1      = (float*)   alloc((size_t)Bc * Nn * 4);
    float*    g2      = (float*)   alloc((size_t)Bc * Nn * 4);

    // Prep
    k_pack_adj<<<2048, 256, 0, stream>>>(adj, adjbits, Bc * Nn * (Nn / 32));
    k_cvt_bf16<<<4096, 256, 0, stream>>>(x, xb, Bc * Nn * NF);
    k_wt  <<<256, 256, 0, stream>>>(W, WT);
    k_wot <<<64, 256, 0, stream>>>(Wo, WoT);

    // Layer 1
    k_wh   <<<Bc * NH * (Nn / 16), 32, 0, stream>>>(xb, WT, WhT);
    k_f1f2 <<<256, 256, 0, stream>>>(WhT, a1, a2, f1, f2);
    k_attn1<<<Bc * (Nn / 16), 256, 0, stream>>>(f1, f2, adjbits, WhT, hb);

    // Output head
    k_wh2  <<<Bc * (Nn / 16), 32, 0, stream>>>(hb, WoT, Wh2, Wh2T);
    k_g    <<<32, 256, 0, stream>>>(Wh2, ao1, ao2, g1, g2);
    k_attn2<<<Bc * (Nn / 16) / 8, 256, 0, stream>>>(g1, g2, adjbits, Wh2T, out);
}
